// ScaledDotProductAttention_59906203844623
// MI455X (gfx1250) — matrix operand order repaired and checked by reading, not compiled, not run on hardware
//
#include <hip/hip_runtime.h>

typedef __attribute__((ext_vector_type(16))) _Float16 v16h;
typedef __attribute__((ext_vector_type(8)))  float    v8f;
typedef unsigned int u32;
typedef __attribute__((ext_vector_type(4))) u32 u32x4;
typedef __attribute__((ext_vector_type(8))) int i32x8;
typedef __attribute__((ext_vector_type(4))) int i32x4;

#define SDIM 2048
#define DDIM 64
#define QT   16
#define NWAVE 8
#define NEG_INF_F (-1e9f)
#define INV_T 0.125f   // 1 / sqrt(64)

// LDS layout in floats (dynamic LDS starts at segment offset 0):
//  sc    [0,     32768)  : 16 x 2048 scores -> probs              (128 KB)
//  kst   [32768, 40960)  : pass1: 8 waves x (16x64) K tiles
//                          end  : 8 waves x (16x64) partial O      (32 KB, time-shared)
//  red   [40960, 41216)  : 16 x 16 reduction partials
//  rmax  [41216, 41232)
//  rsum  [41232, 41248)
//  vst   [41248, 74016)  : 8 waves x 2 x (32x64) V double buffer  (128 KB)
#define SC_OFF   0
#define KST_OFF  32768
#define OACC_OFF 32768
#define RED_OFF  40960
#define RMAX_OFF 41216
#define RSUM_OFF 41232
#define VST_OFF  41248
#define SMEM_FLOATS 74016   // ~289 KB of the 320 KB WGP LDS

// 1-D contiguous Tensor-Data-Mover load: nelem f32 from gaddr -> LDS byte
// offset lds_byte.  D# packing per CDNA5 ISA.
// amdgpu-toolchain (clang-23) 6-arg builtin form.
__device__ __forceinline__ void tdm_load_1d(u32 lds_byte, const float* gaddr, u32 nelem) {
  unsigned long long ga = (unsigned long long)(uintptr_t)gaddr;
  u32x4 g0;
  g0[0] = 1u;                                              // count=1, user mode
  g0[1] = (u32)__builtin_amdgcn_readfirstlane((int)lds_byte);
  g0[2] = (u32)__builtin_amdgcn_readfirstlane((int)(u32)(ga & 0xFFFFFFFFu));
  g0[3] = (u32)__builtin_amdgcn_readfirstlane(
              (int)(((u32)((ga >> 32) & 0x01FFFFFFu)) | (2u << 30)));  // addr[56:32] | type=2
  i32x8 g1;
  g1[0] = (int)(2u << 16);                    // data_size = 4 bytes; mask/flags = 0
  g1[1] = (int)((nelem & 0xFFFFu) << 16);     // tensor_dim0[15:0]  (bits 63:48)
  g1[2] = (int)(((nelem >> 16) & 0xFFFFu) | (1u << 16)); // tensor_dim0[31:16] | tensor_dim1=1
  g1[3] = (int)((nelem & 0xFFFFu) << 16);     // tile_dim0 (bits 127:112)
  g1[4] = 0;                                  // tile_dim1 = tile_dim2 = 0 (unused)
  g1[5] = (int)nelem;                         // tensor_dim0_stride[31:0]
  g1[6] = 0;
  g1[7] = 0;
  i32x4 z4 = {0, 0, 0, 0};
  i32x8 z8 = {0, 0, 0, 0, 0, 0, 0, 0};
  __builtin_amdgcn_tensor_load_to_lds(g0, g1, z4, z4, z8, 0);
}

extern "C" __global__ void attn_fwd_wmma_tdm(const float* __restrict__ q,
                                             const float* __restrict__ kmat,
                                             const float* __restrict__ vmat,
                                             const int*   __restrict__ mask,
                                             float* __restrict__ out,
                                             float* __restrict__ attn) {
  extern __shared__ float smem[];
  float* sc   = smem + SC_OFF;
  float* oacc = smem + OACC_OFF;
  float* red  = smem + RED_OFF;
  float* rmax = smem + RMAX_OFF;
  float* rsum = smem + RSUM_OFF;

  const int tid  = threadIdx.x;
  const int wave = tid >> 5;
  const int lane = tid & 31;
  const int half = lane >> 4;      // 0: lanes 0-15, 1: lanes 16-31
  const int ln   = lane & 15;

  const int bh    = blockIdx.y;    // b*H + h, H == 16
  const int b     = bh >> 4;
  const int qbase = blockIdx.x * QT;

  // ---- Load Q A-fragments (16x32 f16 layout, ISA 7.12.2), pre-scaled by 1/T ----
  const float* qRow = q + (size_t)(bh * SDIM + qbase + ln) * DDIM;
  v16h aq0, aq1;
#pragma unroll
  for (int vv = 0; vv < 8; ++vv) {
    int kOff = ((vv < 4) ? 0 : 16) + (half ? 8 : 0) + 2 * (vv & 3);
    aq0[2*vv]   = (_Float16)(qRow[kOff]          * INV_T);
    aq0[2*vv+1] = (_Float16)(qRow[kOff + 1]      * INV_T);
    aq1[2*vv]   = (_Float16)(qRow[32 + kOff]     * INV_T);
    aq1[2*vv+1] = (_Float16)(qRow[32 + kOff + 1] * INV_T);
  }

  // ---- Pass 1: scores = (Q/T) @ K^T (masked) -> LDS, K tiles staged via TDM ----
  const int NT = SDIM / 16;                       // 128 key tiles
  float* kbuf = smem + KST_OFF + wave * 1024;     // this wave's 16x64 K tile
  const u32 kbuf_byte = (u32)((KST_OFF + wave * 1024) * 4);
  tdm_load_1d(kbuf_byte, kmat + (size_t)(bh * SDIM + wave * 16) * DDIM, 1024);
  for (int t = wave; t < NT; t += NWAVE) {
    __builtin_amdgcn_s_wait_tensorcnt(0);
    v16h bk0, bk1;                                // B: 32x16 f16 layout, from LDS
#pragma unroll
    for (int vv = 0; vv < 8; ++vv) {
      int kk = ((vv < 4) ? 0 : 16) + (half ? 8 : 0) + 2 * (vv & 3);
      bk0[2*vv]   = (_Float16)kbuf[ln * DDIM + kk];
      bk0[2*vv+1] = (_Float16)kbuf[ln * DDIM + kk + 1];
      bk1[2*vv]   = (_Float16)kbuf[ln * DDIM + 32 + kk];
      bk1[2*vv+1] = (_Float16)kbuf[ln * DDIM + 32 + kk + 1];
    }
    // stage next tile while we do the WMMAs (buffer free once DScnt==0)
    if (t + NWAVE < NT) {
      asm volatile("s_wait_dscnt 0" ::: "memory");
      tdm_load_1d(kbuf_byte, kmat + (size_t)(bh * SDIM + (t + NWAVE) * 16) * DDIM, 1024);
    }
    v8f c = {};
    c = __builtin_amdgcn_wmma_f32_16x16x32_f16(false, aq0, false, bk0, (short)0, c, false, false);
    c = __builtin_amdgcn_wmma_f32_16x16x32_f16(false, aq1, false, bk1, (short)0, c, false, false);
    const int key  = t * 16 + ln;
    const int mval = mask[b * SDIM + key];
#pragma unroll
    for (int r = 0; r < 8; ++r) {
      float val = mval ? c[r] : NEG_INF_F;        // C/D: M = r + 8*half, N = ln
      sc[(r + half * 8) * SDIM + key] = val;
    }
  }
  __syncthreads();

  // ---- Row max ----
  {
    const int row = tid & 15, seg = tid >> 4;
    float m = -3.0e38f;
    const float* p = sc + row * SDIM + seg * 128;
    for (int i = 0; i < 128; ++i) m = fmaxf(m, p[i]);
    red[row * 16 + seg] = m;
  }
  __syncthreads();
  if (tid < 16) {
    float m = -3.0e38f;
    for (int s = 0; s < 16; ++s) m = fmaxf(m, red[tid * 16 + s]);
    rmax[tid] = m;
  }
  __syncthreads();
  // ---- Row sum of exp ----
  {
    const int row = tid & 15, seg = tid >> 4;
    const float rm = rmax[row];
    float s = 0.f;
    const float* p = sc + row * SDIM + seg * 128;
    for (int i = 0; i < 128; ++i) s += __expf(p[i] - rm);
    red[row * 16 + seg] = s;
  }
  __syncthreads();
  if (tid < 16) {
    float s = 0.f;
    for (int t = 0; t < 16; ++t) s += red[tid * 16 + t];
    rsum[tid] = 1.0f / s;
  }
  __syncthreads();

  // ---- Normalize: stream attn to global, keep p in LDS for PV GEMM ----
  {
    const int row = tid >> 4, l = tid & 15;
    const float rm = rmax[row], is = rsum[row];
    float* srow = sc + row * SDIM;
    float* arow = attn + ((size_t)bh * SDIM + qbase + row) * SDIM;
    for (int i = 0; i < 128; ++i) {
      const int col = l + i * 16;
      const float pn = __expf(srow[col] - rm) * is;
      srow[col] = pn;
      arow[col] = pn;
    }
  }
  __syncthreads();

  // ---- Pass 2: O = P @ V, V chunks staged via TDM (double-buffered) ----
  v8f acc[4] = {};                                // 4 N-tiles of 16 -> D=64
  const int CPW = (SDIM / 32) / NWAVE;            // 8 K-chunks of 32 per wave
  float* vst = smem + VST_OFF + wave * 4096;      // 2 x 2048-float buffers
  const u32 vst_byte = (u32)((VST_OFF + wave * 4096) * 4);
  tdm_load_1d(vst_byte, vmat + (size_t)(bh * SDIM + wave * CPW * 32) * DDIM, 2048);
  for (int i = 0; i < CPW; ++i) {
    const int kb = (wave * CPW + i) * 32;
    if (i + 1 < CPW) {
      asm volatile("s_wait_dscnt 0" ::: "memory");
      tdm_load_1d(vst_byte + ((u32)((i + 1) & 1)) * 8192u,
                  vmat + (size_t)(bh * SDIM + kb + 32) * DDIM, 2048);
      __builtin_amdgcn_s_wait_tensorcnt(1);       // chunk i complete, i+1 in flight
    } else {
      __builtin_amdgcn_s_wait_tensorcnt(0);
    }
    const float* vb = vst + (i & 1) * 2048;
    v16h fa;                                      // P tile 16x32 as A, from LDS probs
#pragma unroll
    for (int vv = 0; vv < 8; ++vv) {
      int kOff = ((vv < 4) ? 0 : 16) + (half ? 8 : 0) + 2 * (vv & 3);
      fa[2*vv]   = (_Float16)sc[ln * SDIM + kb + kOff];
      fa[2*vv+1] = (_Float16)sc[ln * SDIM + kb + kOff + 1];
    }
#pragma unroll
    for (int nt = 0; nt < 4; ++nt) {
      v16h fb;                                    // V tile 32x16 as B, from staged LDS
#pragma unroll
      for (int vv = 0; vv < 8; ++vv) {
        int kk = ((vv < 4) ? 0 : 16) + (half ? 8 : 0) + 2 * (vv & 3);
        fb[2*vv]   = (_Float16)vb[kk * DDIM + nt * 16 + ln];
        fb[2*vv+1] = (_Float16)vb[(kk + 1) * DDIM + nt * 16 + ln];
      }
      acc[nt] = __builtin_amdgcn_wmma_f32_16x16x32_f16(false, fa, false, fb, (short)0, acc[nt], false, false);
    }
  }
  // store per-wave partial O to LDS (kst region is dead now)
#pragma unroll
  for (int nt = 0; nt < 4; ++nt)
#pragma unroll
    for (int r = 0; r < 8; ++r) {
      const int m  = r + half * 8;
      const int dc = nt * 16 + ln;
      oacc[wave * 1024 + m * 64 + dc] = acc[nt][r];
    }
  __syncthreads();

  // ---- Cross-wave reduce + write output ----
  for (int idx = tid; idx < 1024; idx += 256) {
    float s = 0.f;
#pragma unroll
    for (int w = 0; w < NWAVE; ++w) s += oacc[w * 1024 + idx];
    const int m = idx >> 6, dc = idx & 63;
    out[((size_t)bh * SDIM + qbase + m) * DDIM + dc] = s;
  }
}

extern "C" void kernel_launch(void* const* d_in, const int* in_sizes, int n_in,
                              void* d_out, int out_size, void* d_ws, size_t ws_size,
                              hipStream_t stream) {
  (void)in_sizes; (void)n_in; (void)d_ws; (void)ws_size; (void)out_size;
  const float* q    = (const float*)d_in[0];
  const float* kmat = (const float*)d_in[1];
  const float* vmat = (const float*)d_in[2];
  const int*   mask = (const int*)d_in[3];
  float* out  = (float*)d_out;
  float* attn = out + (size_t)2 * 16 * SDIM * DDIM;   // output first, then attn

  dim3 grid(SDIM / QT, 2 * 16);    // 128 q-tiles x (B*H)=32
  dim3 block(256);                 // 8 wave32
  size_t smem = SMEM_FLOATS * sizeof(float);
  attn_fwd_wmma_tdm<<<grid, block, smem, stream>>>(q, kmat, vmat, mask, out, attn);
}
